// ClipClap_77438260347097
// MI455X (gfx1250) — matrix operand order repaired and checked by reading, not compiled, not run on hardware
//
#include <hip/hip_runtime.h>
#include <math.h>

// ---------------------------------------------------------------------------
// MI455X / gfx1250, wave32. Matmuls via v_wmma_f32_16x16x32_f16; K/V tiles
// staged to LDS with the Tensor Data Mover (tensor_load_to_lds + TENSORcnt).
// Problem: L=S=49, B=2048, D=256, H=8, D_K=32  (D_K == WMMA K -> 1 k-step)
// ---------------------------------------------------------------------------

typedef __attribute__((ext_vector_type(16))) _Float16     v16h;
typedef __attribute__((ext_vector_type(8)))  float        v8f;
typedef __attribute__((ext_vector_type(4)))  unsigned int v4u;
typedef __attribute__((ext_vector_type(8)))  int          v8i;
typedef __attribute__((ext_vector_type(4)))  int          v4i;

#define WMMA_F16(a, b, c) \
  __builtin_amdgcn_wmma_f32_16x16x32_f16(false, (a), false, (b), (short)0, (c), false, false)

#define NEG_INF (-__builtin_inff())

// ---- Tensor Data Mover: 1-D contiguous tile, global -> LDS ---------------
// D# per CDNA5 ISA ch.8: group0 {count=1, lds_addr, global_addr, type=2},
// group1 {data_size=2B, tensor_dim0=tile_dim0=nelem, dims1=1, stride0=nelem}.
__device__ __forceinline__ void tdm_load_2b(unsigned lds_off, const void* gptr,
                                            unsigned nelem /* 2-byte elems */) {
  unsigned long long ga = (unsigned long long)(uintptr_t)gptr;
  v4u g0;
  g0[0] = 1u;                                            // count=1, user mode
  g0[1] = lds_off;                                       // lds_addr (bytes)
  g0[2] = (unsigned)(ga & 0xffffffffu);                  // global_addr[31:0]
  g0[3] = (unsigned)((ga >> 32) & 0x01ffffffu) | 0x80000000u; // [56:32] | type=2
  v8i g1;
  g1[0] = 1 << 16;                                       // data_size=1 (2B)
  g1[1] = (int)(nelem << 16);                            // tensor_dim0 lo16
  g1[2] = 1 << 16;                                       // tensor_dim1 = 1
  g1[3] = (int)(nelem << 16);                            // tile_dim0
  g1[4] = 1;                                             // tile_dim1 = 1
  g1[5] = (int)nelem;                                    // tensor_dim0_stride
  g1[6] = 0;
  g1[7] = 0;
  v4i z = {0, 0, 0, 0};
#if defined(__clang_major__) && (__clang_major__ >= 23)
  v8i z8 = {0, 0, 0, 0, 0, 0, 0, 0};
  __builtin_amdgcn_tensor_load_to_lds(g0, g1, z, z, z8, 0);
#else
  __builtin_amdgcn_tensor_load_to_lds(g0, g1, z, z, 0);
#endif
}

// ---- Fragment loaders (contiguous k = 16*h + s in A and B, wave32) --------------
__device__ __forceinline__ v16h load_a_f32(const float* __restrict__ M, int ld,
                                           int row0, int k0) {
  int lane = threadIdx.x & 31;
  int m  = row0 + (lane & 15);
  int kb = (lane & 16) ? 16 : 0;
  const float* p0 = M + (size_t)m * ld + k0 + kb;
  const float* p1 = p0 + 8;
  v16h a;
#pragma unroll
  for (int e = 0; e < 8; ++e) {
    a[e]     = (_Float16)p0[e];
    a[e + 8] = (_Float16)p1[e];
  }
  return a;
}

__device__ __forceinline__ v16h load_a_f16(const _Float16* M, int ld,
                                           int row0, int k0, int maxrow) {
  int lane = threadIdx.x & 31;
  int m = row0 + (lane & 15);
  if (m > maxrow) m = maxrow;            // clamp into valid rows (padding-safe)
  int kb = (lane & 16) ? 16 : 0;
  const _Float16* p0 = M + (size_t)m * ld + k0 + kb;
  const _Float16* p1 = p0 + 8;
  v16h a;
#pragma unroll
  for (int e = 0; e < 8; ++e) {
    a[e]     = p0[e];
    a[e + 8] = p1[e];
  }
  return a;
}

// B (32x16): lanes 0-15 col N=lane hold K=0..15; lanes 16-31 hold K=16..31.
// "Transposed" loader: B[k][n] = M[n0+n][k0+k]  (computes A * M^T)
__device__ __forceinline__ v16h load_bT_f16(const _Float16* M, int ld,
                                            int n0, int k0, int maxrow) {
  int lane = threadIdx.x & 31;
  int n = n0 + (lane & 15);
  if (n > maxrow) n = maxrow;
  int kb = (lane & 16) ? 16 : 0;
  const _Float16* p = M + (size_t)n * ld + k0 + kb;
  v16h b;
#pragma unroll
  for (int e = 0; e < 16; ++e) b[e] = p[e];
  return b;
}

// Direct B: B[k][n] = M[k0+k][n0+n]   (for P*V from LDS), row clamped
__device__ __forceinline__ v16h load_b_direct_f16(const _Float16* M, int ld,
                                                  int k0, int n0, int maxrow) {
  int lane = threadIdx.x & 31;
  int n  = n0 + (lane & 15);
  int kb = (lane & 16) ? 16 : 0;
  v16h b;
#pragma unroll
  for (int e = 0; e < 16; ++e) {
    int kr = k0 + kb + e;
    if (kr > maxrow) kr = maxrow;
    b[e] = M[(size_t)kr * ld + n];
  }
  return b;
}

// C/D (16x16 f32, 8 VGPRs): lanes 0-15 N=lane M=r; lanes 16-31 N=lane-16 M=8+r
__device__ __forceinline__ void store_c_lds(float* M, int ld, int m0, int n0, v8f c) {
  int lane = threadIdx.x & 31;
  int n  = n0 + (lane & 15);
  int mb = (lane & 16) ? 8 : 0;
#pragma unroll
  for (int r = 0; r < 8; ++r) M[(m0 + mb + r) * ld + n] = c[r];
}

// ---------------------------------------------------------------------------
// Kernel 0: one-time f32 -> f16 weight conversion (in_proj 768x256, out 256x256)
// ---------------------------------------------------------------------------
__global__ void ClipClap_wcvt(const float* __restrict__ w0, const float* __restrict__ w1,
                              _Float16* __restrict__ o0, _Float16* __restrict__ o1) {
  int i = blockIdx.x * 256 + threadIdx.x;
  if (i < 768 * 256) o0[i] = (_Float16)w0[i];
  if (i < 256 * 256) o1[i] = (_Float16)w1[i];
}

// ---------------------------------------------------------------------------
// Kernel 1: positional key tables  k_pos{y,x}[h][p][dk] (f16, p padded to 48)
// ---------------------------------------------------------------------------
__global__ void ClipClap_pos(const float* __restrict__ posy, const float* __restrict__ posx,
                             const float* __restrict__ Wk, _Float16* __restrict__ PY,
                             _Float16* __restrict__ PX) {
  int idx = blockIdx.x * blockDim.x + threadIdx.x;
  const int TOT = 2 * 8 * 48 * 32;
  if (idx >= TOT) return;
  int which = idx / (8 * 48 * 32);
  int rem   = idx % (8 * 48 * 32);
  int h  = rem / (48 * 32);
  int pr = (rem >> 5) % 48;
  int dk = rem & 31;
  int d  = h * 32 + dk;
  float acc = 0.f;
  if (pr < 41) {
    const float* tab = (which ? posx : posy) + (size_t)pr * 128;
    const float* w   = Wk + (size_t)d * 256 + (which ? 128 : 0);
#pragma unroll 4
    for (int f = 0; f < 128; ++f) acc += tab[f] * w[f];
  }
  _Float16* o = which ? PX : PY;
  o[((size_t)h * 48 + pr) * 32 + dk] = (_Float16)acc;
}

// ---------------------------------------------------------------------------
// Kernel 2: QKV projection GEMM. out[b][h][l][dk] = (X @ W^T + bias) * scale
// grid.x = 6272 row tiles (16 rows = one l, 16 consecutive b), 4 waves,
// wave wv covers cols [64*wv, 64*wv+64). Weights pre-converted to f16.
// ---------------------------------------------------------------------------
__global__ __launch_bounds__(128) void ClipClap_qkv_proj(
    const float* __restrict__ X, const _Float16* __restrict__ W,
    const float* __restrict__ bias, _Float16* __restrict__ out, float scale) {
  int row0 = blockIdx.x * 16;
  int wv   = threadIdx.x >> 5;
  int lane = threadIdx.x & 31;

  v8f acc[4] = {};
#pragma unroll
  for (int k0 = 0; k0 < 256; k0 += 32) {
    if (k0 == 0)
      __builtin_prefetch(X + (size_t)(row0 + (lane & 15)) * 256 + 128, 0, 0);
    v16h a = load_a_f32(X, 256, row0, k0);
#pragma unroll
    for (int ct = 0; ct < 4; ++ct) {
      v16h b = load_bT_f16(W, 256, wv * 64 + ct * 16, k0, 1 << 30);
      acc[ct] = WMMA_F16(a, b, acc[ct]);
    }
  }

  int l     = row0 >> 11;          // rows in a tile share l (2048 % 16 == 0)
  int bbase = row0 & 2047;
  int mb    = (lane & 16) ? 8 : 0;
#pragma unroll
  for (int ct = 0; ct < 4; ++ct) {
    int col = wv * 64 + ct * 16 + (lane & 15);
    int h = col >> 5, dk = col & 31;
    float bv = bias[col];
#pragma unroll
    for (int r = 0; r < 8; ++r) {
      int bidx = bbase + mb + r;
      float val = (acc[ct][r] + bv) * scale;
      out[(((size_t)bidx * 8 + h) * 49 + l) * 32 + dk] = (_Float16)val;
    }
  }
}

// ---------------------------------------------------------------------------
// Kernel 3: fused attention per (b,h). 4 waves; wave lt owns l-tile lt.
// K,V tiles (49x32 f16, contiguous 3136 B) DMA'd to LDS by the TDM.
// S = Q K^T, Py = Q posY^T, Px = Q posX^T -> LDS f32; gather + softmax -> P f16
// in LDS; O = P V -> f16 to AO[(l*B+b)*256 + h*32 + dk].
// ---------------------------------------------------------------------------
__global__ __launch_bounds__(128) void ClipClap_attn(
    const _Float16* __restrict__ Qg, const _Float16* __restrict__ Kg,
    const _Float16* __restrict__ Vg, const _Float16* __restrict__ PYg,
    const _Float16* __restrict__ PXg, const unsigned char* __restrict__ maskg,
    _Float16* __restrict__ AOg) {
  __shared__ float    Ssc[64 * 64];
  __shared__ float    PYs[64 * 48];
  __shared__ float    PXs[64 * 48];
  __shared__ _Float16 Ps[64 * 64];
  __shared__ __align__(16) _Float16 Ks[49 * 32];
  __shared__ __align__(16) _Float16 Vs[49 * 32];
  __shared__ int      yys[49], xxs[49];
  __shared__ float    mneg[64];

  int bh = blockIdx.x;
  int b = bh >> 3, h = bh & 7;
  int tid = threadIdx.x, lane = tid & 31, lt = tid >> 5;

  const _Float16* Qb  = Qg + (size_t)bh * 49 * 32;
  const _Float16* Kb  = Kg + (size_t)bh * 49 * 32;
  const _Float16* Vb  = Vg + (size_t)bh * 49 * 32;
  const _Float16* PYb = PYg + (size_t)h * 48 * 32;
  const _Float16* PXb = PXg + (size_t)h * 48 * 32;

  // --- TDM: DMA K and V tiles into LDS (wave 0 issues; EXEC ignored) ------
  if (lt == 0) {
    tdm_load_2b((unsigned)(uintptr_t)(void*)&Ks[0], Kb, 49 * 32);
    tdm_load_2b((unsigned)(uintptr_t)(void*)&Vs[0], Vb, 49 * 32);
  }

  const unsigned char* mrow = maskg + (size_t)b * 49;
  if (tid < 49) {
    int r = tid / 7, c = tid - r * 7;
    int sy = 0, sx = 0;
#pragma unroll
    for (int rr = 0; rr < 7; ++rr) if (rr <= r) sy += mrow[rr * 7 + c] ? 0 : 1;
#pragma unroll
    for (int cc = 0; cc < 7; ++cc) if (cc <= c) sx += mrow[r * 7 + cc] ? 0 : 1;
    yys[tid] = sy;
    xxs[tid] = sx;
  }
  if (tid < 64) mneg[tid] = (tid < 49 && mrow[tid] == 0) ? 0.0f : NEG_INF;

  if (lt == 0) __builtin_amdgcn_s_wait_tensorcnt(0);   // K/V resident in LDS
  __syncthreads();

  // --- scores + positional biases (K-dim = D_K = 32 -> single WMMA each) ---
  v16h aq = load_a_f16(Qb, 32, lt * 16, 0, 48);
#pragma unroll
  for (int st = 0; st < 4; ++st) {
    v16h bk = load_bT_f16(Ks, 32, st * 16, 0, 48);
    v8f  c  = {};
    c = WMMA_F16(aq, bk, c);
    store_c_lds(Ssc, 64, lt * 16, st * 16, c);
  }
#pragma unroll
  for (int pt = 0; pt < 3; ++pt) {
    v16h by = load_bT_f16(PYb, 32, pt * 16, 0, 47);
    v8f  cy = {};
    cy = WMMA_F16(aq, by, cy);
    store_c_lds(PYs, 48, lt * 16, pt * 16, cy);
    v16h bx = load_bT_f16(PXb, 32, pt * 16, 0, 47);
    v8f  cx = {};
    cx = WMMA_F16(aq, bx, cx);
    store_c_lds(PXs, 48, lt * 16, pt * 16, cx);
  }

  // --- per-row gather + softmax; wave lt handles its own 16 rows ----------
  for (int r = 0; r < 16; ++r) {
    int l = lt * 16 + r;
    float p0 = 0.f, p1 = 0.f;
    if (l < 49) {
      int s0 = lane, s1 = lane + 32;
      float v0 = Ssc[l * 64 + s0] + mneg[s0];
      int iy = yys[l] - yys[s0] + 20;
      int ix = xxs[l] - xxs[s0] + 20;
      v0 += PYs[l * 48 + iy] + PXs[l * 48 + ix];
      float v1 = NEG_INF;
      if (s1 < 49) {
        float t = Ssc[l * 64 + s1] + mneg[s1];
        int iy1 = yys[l] - yys[s1] + 20;
        int ix1 = xxs[l] - xxs[s1] + 20;
        v1 = t + PYs[l * 48 + iy1] + PXs[l * 48 + ix1];
      }
      float mx = fmaxf(v0, v1);
#pragma unroll
      for (int off = 16; off > 0; off >>= 1) mx = fmaxf(mx, __shfl_xor(mx, off, 32));
      float e0 = (v0 > NEG_INF) ? __expf(v0 - mx) : 0.f;
      float e1 = (v1 > NEG_INF) ? __expf(v1 - mx) : 0.f;
      float sum = e0 + e1;
#pragma unroll
      for (int off = 16; off > 0; off >>= 1) sum += __shfl_xor(sum, off, 32);
      float inv = 1.0f / fmaxf(sum, 1e-20f);
      p0 = e0 * inv;
      p1 = e1 * inv;
    }
    Ps[l * 64 + lane]      = (_Float16)p0;   // padding rows/cols forced to 0
    Ps[l * 64 + lane + 32] = (_Float16)p1;
  }

  // --- O = P * V  (two k-steps over s, two 16-col d-tiles) ----------------
  v16h a0 = load_a_f16(Ps, 64, lt * 16, 0, 63);
  v16h a1 = load_a_f16(Ps, 64, lt * 16, 32, 63);
  v8f  o0 = {}, o1 = {};
  {
    v16h b00 = load_b_direct_f16(Vs, 32, 0, 0, 48);
    o0 = WMMA_F16(a0, b00, o0);
    v16h b10 = load_b_direct_f16(Vs, 32, 32, 0, 48);
    o0 = WMMA_F16(a1, b10, o0);
    v16h b01 = load_b_direct_f16(Vs, 32, 0, 16, 48);
    o1 = WMMA_F16(a0, b01, o1);
    v16h b11 = load_b_direct_f16(Vs, 32, 32, 16, 48);
    o1 = WMMA_F16(a1, b11, o1);
  }

  int mb = (lane & 16) ? 8 : 0;
  int n  = lane & 15;
#pragma unroll
  for (int r = 0; r < 8; ++r) {
    int l = lt * 16 + mb + r;
    if (l < 49) {
      size_t rowb = ((size_t)l * 2048 + b) * 256 + h * 32;
      AOg[rowb + n]      = (_Float16)o0[r];
      AOg[rowb + 16 + n] = (_Float16)o1[r];
    }
  }
}

// ---------------------------------------------------------------------------
// Kernel 4: output projection. out = AO(f16) @ out_w^T + out_b  (f32 result)
// ---------------------------------------------------------------------------
__global__ __launch_bounds__(128) void ClipClap_out_proj(
    const _Float16* __restrict__ A, const _Float16* __restrict__ W,
    const float* __restrict__ bias, float* __restrict__ out) {
  int row0 = blockIdx.x * 16;
  int wv   = threadIdx.x >> 5;
  int lane = threadIdx.x & 31;

  v8f acc[4] = {};
#pragma unroll
  for (int k0 = 0; k0 < 256; k0 += 32) {
    v16h a = load_a_f16(A, 256, row0, k0, 100351);
#pragma unroll
    for (int ct = 0; ct < 4; ++ct) {
      v16h b = load_bT_f16(W, 256, wv * 64 + ct * 16, k0, 1 << 30);
      acc[ct] = WMMA_F16(a, b, acc[ct]);
    }
  }

  int mb = (lane & 16) ? 8 : 0;
#pragma unroll
  for (int ct = 0; ct < 4; ++ct) {
    int col = wv * 64 + ct * 16 + (lane & 15);
    float bv = bias[col];
#pragma unroll
    for (int r = 0; r < 8; ++r)
      out[(size_t)(row0 + mb + r) * 256 + col] = acc[ct][r] + bv;
  }
}

// ---------------------------------------------------------------------------
extern "C" void kernel_launch(void* const* d_in, const int* in_sizes, int n_in,
                              void* d_out, int out_size, void* d_ws, size_t ws_size,
                              hipStream_t stream) {
  (void)in_sizes; (void)n_in; (void)out_size; (void)ws_size;

  const float*         query = (const float*)d_in[0];
  const float*         key   = (const float*)d_in[1];
  const float*         value = (const float*)d_in[2];
  const unsigned char* mask  = (const unsigned char*)d_in[3];
  const float*         in_w  = (const float*)d_in[4];   // [768][256]
  const float*         in_b  = (const float*)d_in[5];   // [768]
  const float*         out_w = (const float*)d_in[6];   // [256][256]
  const float*         out_b = (const float*)d_in[7];   // [256]
  const float*         pos_x = (const float*)d_in[8];   // [41][128]
  const float*         pos_y = (const float*)d_in[9];   // [41][128]
  float*               out   = (float*)d_out;

  const size_t nQ = (size_t)2048 * 8 * 49 * 32;   // 25,690,112 halfs
  _Float16* Q    = (_Float16*)d_ws;
  _Float16* K    = Q + nQ;
  _Float16* V    = K + nQ;
  _Float16* AO   = V + nQ;                        // [100352][256] f16
  _Float16* PY   = AO + (size_t)100352 * 256;
  _Float16* PX   = PY + 8 * 48 * 32;
  _Float16* WINh = PX + 8 * 48 * 32;              // [768][256] f16
  _Float16* WOUTh = WINh + (size_t)768 * 256;     // [256][256] f16

  const float scale = 0.17677669529663689f;       // 1/sqrt(D_K)

  ClipClap_wcvt<<<768, 256, 0, stream>>>(in_w, out_w, WINh, WOUTh);

  ClipClap_pos<<<(2 * 8 * 48 * 32 + 255) / 256, 256, 0, stream>>>(
      pos_y, pos_x, in_w + 65536, PY, PX);

  ClipClap_qkv_proj<<<6272, 128, 0, stream>>>(query, WINh,          in_b,       Q, scale);
  ClipClap_qkv_proj<<<6272, 128, 0, stream>>>(key,   WINh + 65536,  in_b + 256, K, 1.0f);
  ClipClap_qkv_proj<<<6272, 128, 0, stream>>>(value, WINh + 131072, in_b + 512, V, 1.0f);

  ClipClap_attn<<<16384, 128, 0, stream>>>(Q, K, V, PY, PX, mask, AO);

  ClipClap_out_proj<<<6272, 128, 0, stream>>>(AO, WOUTh, out_b, out);
}
